// Interstellar_66889820668497
// MI455X (gfx1250) — matrix operand order repaired and checked by reading, not compiled, not run on hardware
//
#include <hip/hip_runtime.h>

// ---------------------------------------------------------------------------
// MI455X (gfx1250) implementation.
// Roofline: 18 GEMMs of 32768x512x512 (~310 GFLOP) dominate; HBM traffic is
// ~0.6 GB (~26us @ 23.3 TB/s). GEMMs run in bf16 WMMA (select outputs feed an
// argmax only, so bf16 is numerically safe); everything value-carrying stays
// fp32 elementwise. GEMM staging uses CDNA5 async global->LDS DMA
// (global_load_async_to_lds_b128 + s_wait_asynccnt) with double-buffered
// LDS tiles so the WMMAs of chunk k overlap the DMA of chunk k+1.
// ---------------------------------------------------------------------------

typedef __bf16 bf16_t;
typedef __attribute__((ext_vector_type(16))) __bf16        v16bf;
typedef __attribute__((ext_vector_type(8)))  float         v8f;
typedef __attribute__((ext_vector_type(4)))  unsigned int  u32x4;

constexpr int    kH    = 512;
constexpr int    kHB   = 256;              // half of H (complex split)
constexpr int    kB    = 8192;
constexpr int    kT    = 3;
constexpr size_t kBH   = (size_t)kB * kH;  // 4,194,304
constexpr int    kRows = 4 * kB;           // 32768 GEMM rows
constexpr size_t kMat  = (size_t)kRows * kH;

union Frag { u32x4 u[2]; v16bf v; };

// CDNA5 async global->LDS copy (ASYNCcnt-tracked), see CDNA5 ISA
__device__ inline void async_ld_b128(unsigned int lds_byte, unsigned long long gaddr) {
  asm volatile("global_load_async_to_lds_b128 %0, %1, off"
               :: "v"(lds_byte), "v"(gaddr) : "memory");
}
__device__ inline void wait_async0() {
  asm volatile("s_wait_asynccnt 0" ::: "memory");
}

// ---------------------------------------------------------------------------
// Weight transpose + bf16 convert: Wt[w][n][k] = (bf16) W_w[k][n]
// ---------------------------------------------------------------------------
__global__ __launch_bounds__(256) void conv_weights(
    const float* __restrict__ w0, const float* __restrict__ w1,
    const float* __restrict__ w2, const float* __restrict__ w3,
    const float* __restrict__ w4, const float* __restrict__ w5,
    bf16_t* __restrict__ Wt) {
  const float* Ws[6] = {w0, w1, w2, w3, w4, w5};
  int w = blockIdx.y;
  int idx = blockIdx.x * 256 + threadIdx.x;  // 0..262143
  int n = idx >> 9, k = idx & 511;
  Wt[(size_t)w * kH * kH + (size_t)n * kH + k] = (bf16_t)Ws[w][(size_t)k * kH + n];
}

// ---------------------------------------------------------------------------
// Embedding gather: s_raw/r_raw (t,b,h), target half of d_out, h0 carry.
// ---------------------------------------------------------------------------
__global__ __launch_bounds__(256) void gather_embed(
    const int* __restrict__ seqs,
    const float* __restrict__ sub_e, const float* __restrict__ rel_e,
    const float* __restrict__ obj_e,
    float* __restrict__ s_raw, float* __restrict__ r_raw,
    float* __restrict__ hbuf, float* __restrict__ target_out) {
  size_t idx = (size_t)blockIdx.x * 256 + threadIdx.x;  // t*BH + b*H + h
  int t = (int)(idx / kBH);
  size_t rem = idx - (size_t)t * kBH;
  int b = (int)(rem >> 9);
  int h = (int)(rem & 511);
  int si = seqs[b * 7 + 2 * t];
  int ri = seqs[b * 7 + 2 * t + 1];
  int oi = seqs[b * 7 + 2 * t + 2];
  float sv = sub_e[(size_t)si * kH + h];
  s_raw[idx] = sv;
  r_raw[idx] = rel_e[(size_t)ri * kH + h];
  target_out[idx] = obj_e[(size_t)oi * kH + h];
  if (t == 0) hbuf[rem] = sv;  // h0 = raw s_emb[:,0,:]
}

// ---------------------------------------------------------------------------
// Batchnorm stats, two pass: partial sums per (rowgroup,col), then finalize.
// ---------------------------------------------------------------------------
__global__ void stats_pass1(const float* __restrict__ X, int R,
                            float* __restrict__ partial) {
  int tx = threadIdx.x;            // 0..63 (col in chunk)
  int ty = threadIdx.y;            // 0..3
  int col = blockIdx.x * 64 + tx;  // 8 chunks cover 512
  int g = blockIdx.y;              // 0..31 row groups
  float s = 0.f, sq = 0.f;
  for (int r = g * 4 + ty; r < R; r += 128) {
    float v = X[(size_t)r * kH + col];
    s += v; sq += v * v;
  }
  __shared__ float sh[4][64][2];
  sh[ty][tx][0] = s; sh[ty][tx][1] = sq;
  __syncthreads();
  if (ty == 0) {
    for (int i = 1; i < 4; ++i) { s += sh[i][tx][0]; sq += sh[i][tx][1]; }
    partial[((size_t)g * kH + col) * 2 + 0] = s;
    partial[((size_t)g * kH + col) * 2 + 1] = sq;
  }
}

__global__ void stats_fin(const float* __restrict__ partial, int R,
                          float* __restrict__ mean, float* __restrict__ rstd) {
  int col = blockIdx.x * blockDim.x + threadIdx.x;
  if (col >= kH) return;
  float s = 0.f, sq = 0.f;
  for (int g = 0; g < 32; ++g) {
    s  += partial[((size_t)g * kH + col) * 2 + 0];
    sq += partial[((size_t)g * kH + col) * 2 + 1];
  }
  float m = s / (float)R;
  float var = sq / (float)R - m * m;
  mean[col] = m;
  rstd[col] = rsqrtf(var + 1e-5f);
}

// Normalize in place, per-time-step stats (grid.y = t)
__global__ __launch_bounds__(256) void bn_apply3(
    float* __restrict__ data, const float* __restrict__ mean3,
    const float* __restrict__ rstd3, const float* __restrict__ g,
    const float* __restrict__ beta) {
  int t = blockIdx.y;
  size_t i = (size_t)blockIdx.x * 256 + threadIdx.x;  // < kBH
  int h = (int)(i & 511);
  size_t off = (size_t)t * kBH + i;
  data[off] = g[h] * (data[off] - mean3[t * kH + h]) * rstd3[t * kH + h] + beta[h];
}

// Final bn3 -> d_out first half
__global__ __launch_bounds__(256) void bn_out(
    const float* __restrict__ outs, const float* __restrict__ mean,
    const float* __restrict__ rstd, const float* __restrict__ g,
    const float* __restrict__ beta, float* __restrict__ out) {
  size_t i = (size_t)blockIdx.x * 256 + threadIdx.x;  // < 3*kBH
  int h = (int)(i & 511);
  out[i] = g[h] * (outs[i] - mean[h]) * rstd[h] + beta[h];
}

// ---------------------------------------------------------------------------
// Step pre: op1 = sigmoid(complx(st,h)); stack_out = complx([st,h,op1,0], rt)
// Writes fp32 stack_out + bf16 GEMM operand; resets argmax keys.
// One thread per (b, j<256) handles both complex halves.
// ---------------------------------------------------------------------------
__global__ __launch_bounds__(256) void step_pre(
    const float* __restrict__ st, const float* __restrict__ rt,
    const float* __restrict__ hbuf, float* __restrict__ op1,
    float* __restrict__ so, bf16_t* __restrict__ Xb,
    unsigned long long* __restrict__ keys) {
  size_t id = (size_t)blockIdx.x * 256 + threadIdx.x;  // b*256 + j
  int j = (int)(id & 255);
  size_t b = id >> 8;
  size_t base = b * kH;
  float s1 = st[base + j],   s2 = st[base + kHB + j];
  float h1 = hbuf[base + j], h2 = hbuf[base + kHB + j];
  float r1 = rt[base + j],   r2 = rt[base + kHB + j];
  float o_re = s1 * h1 - s2 * h2;
  float o_im = s1 * h2 + s2 * h1;
  float p1 = 1.f / (1.f + __expf(-o_re));
  float p2 = 1.f / (1.f + __expf(-o_im));
  op1[base + j] = p1; op1[base + kHB + j] = p2;
  float xr[4] = {s1, h1, p1, 0.f};
  float xi[4] = {s2, h2, p2, 0.f};
  for (int s = 0; s < 4; ++s) {
    float re = xr[s] * r1 - xi[s] * r2;
    float im = xr[s] * r2 + xi[s] * r1;
    size_t o = (size_t)s * kBH + base;
    so[o + j] = re;        so[o + kHB + j] = im;
    Xb[o + j] = (bf16_t)re; Xb[o + kHB + j] = (bf16_t)im;
  }
  if (id == 0) { keys[0] = 0ull; keys[1] = 0ull; }
}

// ---------------------------------------------------------------------------
// WMMA GEMM: Y[w] (32768x512, bf16) = X (32768x512 bf16) * Wt[w]^T
// Wt stored n-major (k contiguous). Block tile 128x64, 8 waves, each wave
// owns 2x2 16x16 accumulators. K is staged through DOUBLE-BUFFERED LDS tiles
// via async global->LDS DMA: while the WMMAs consume buffer `cur`, the DMA
// for chunk kc+32 fills buffer `cur^1`; s_wait_asynccnt 0 + barrier closes
// each iteration. 80-byte padded LDS rows -> conflict-free ds_read_b128
// fragment loads. grid.z selects weight matrix.
// ---------------------------------------------------------------------------
__global__ __launch_bounds__(256) void gemm_qkv(
    const bf16_t* __restrict__ X, const bf16_t* __restrict__ Wt,
    bf16_t* __restrict__ Y) {
  const int w = blockIdx.z;
  const bf16_t* Wm = Wt + (size_t)w * kH * kH;
  bf16_t* Ym = Y + (size_t)w * kMat;
  const int m0 = blockIdx.x * 128;
  const int n0 = blockIdx.y * 64;
  const int tid = threadIdx.x;
  const int wave = tid >> 5, lane = tid & 31;
  const int mw = (wave & 3) * 32;   // 4 wave-rows cover 128
  const int nw = (wave >> 2) * 32;  // 2 wave-cols cover 64
  const int hi16 = lane >> 4;       // 0 or 1 (half-wave)
  const int l15 = lane & 15;

  __shared__ u32x4 ldsA[2][128 * 5];  // 128 rows x 64B, +16B pad (80B stride)
  __shared__ u32x4 ldsB[2][64 * 5];   // 64 rows  x 64B, +16B pad

  v8f acc[2][2] = {};

  const int ra = tid >> 1, ha = tid & 1;  // A staging: row, 32B half
  const int rb = tid >> 2, qb = tid & 3;  // B staging: row, 16B quarter

  // Issue the async DMA for K-chunk kc into LDS buffer `buf`.
  // A: each thread moves 32B (2 x b128); B: each thread moves 16B.
  auto stage = [&](int buf, int kc) {
    unsigned int a_off =
        (unsigned int)(uintptr_t)(&ldsA[buf][ra * 5 + ha * 2]);
    unsigned long long ga = (unsigned long long)(uintptr_t)(
        X + (size_t)(m0 + ra) * kH + kc + ha * 16);
    async_ld_b128(a_off, ga);
    async_ld_b128(a_off + 16u, ga + 16ull);
    unsigned int b_off =
        (unsigned int)(uintptr_t)(&ldsB[buf][rb * 5 + qb]);
    unsigned long long gb = (unsigned long long)(uintptr_t)(
        Wm + (size_t)(n0 + rb) * kH + kc + qb * 8);
    async_ld_b128(b_off, gb);
  };

  stage(0, 0);
  wait_async0();
  __syncthreads();

  for (int kc = 0; kc < kH; kc += 32) {
    const int cur = (kc >> 5) & 1;
    if (kc + 32 < kH) stage(cur ^ 1, kc + 32);  // prefetch next chunk

    Frag fa[2], fb[2];
    for (int mi = 0; mi < 2; ++mi) {
      int r = mw + mi * 16 + l15;
      fa[mi].u[0] = ldsA[cur][r * 5 + hi16];      // K 0..7  (or 8..15)
      fa[mi].u[1] = ldsA[cur][r * 5 + 2 + hi16];  // K 16..23 (or 24..31)
    }
    for (int ni = 0; ni < 2; ++ni) {
      int n = nw + ni * 16 + l15;
      fb[ni].u[0] = ldsB[cur][n * 5 + hi16];      // K 0..7  (or 8..15)
      fb[ni].u[1] = ldsB[cur][n * 5 + 2 + hi16];  // K 16..23 (or 24..31)
    }
    for (int mi = 0; mi < 2; ++mi)
      for (int ni = 0; ni < 2; ++ni)
        acc[mi][ni] = __builtin_amdgcn_wmma_f32_16x16x32_bf16(
            false, fa[mi].v, false, fb[ni].v, (short)0, acc[mi][ni],
            false, false);

    wait_async0();     // prefetch DMA complete (no-op on last iter)
    __syncthreads();   // all waves done reading `cur` / writing `cur^1`
  }

  // C/D layout: lane<16 -> rows M=0..7 in acc[0..7], lane>=16 -> M=8..15
  for (int mi = 0; mi < 2; ++mi)
    for (int ni = 0; ni < 2; ++ni) {
      int rbase = m0 + mw + mi * 16 + hi16 * 8;
      int col = n0 + nw + ni * 16 + l15;
      for (int i = 0; i < 8; ++i)
        Ym[(size_t)(rbase + i) * kH + col] = (bf16_t)acc[mi][ni][i];
    }
}

// ---------------------------------------------------------------------------
// Per-b 4x4 attention + global argmax (packed value<<32 | ~flatidx, atomicMax)
// One wave per b; lanes stripe the 512 feature dim.
// ---------------------------------------------------------------------------
__device__ inline unsigned sortable_u32(float f) {
  unsigned u = __float_as_uint(f);
  return (u & 0x80000000u) ? ~u : (u | 0x80000000u);
}

__global__ __launch_bounds__(256) void attn_select(
    const bf16_t* __restrict__ QKV, unsigned long long* __restrict__ keys) {
  const int wave = threadIdx.x >> 5, lane = threadIdx.x & 31;
  const int b = blockIdx.x * 8 + wave;
  for (int head = 0; head < 2; ++head) {
    const bf16_t* Q = QKV + (size_t)(head * 3 + 0) * kMat;
    const bf16_t* K = QKV + (size_t)(head * 3 + 1) * kMat;
    const bf16_t* V = QKV + (size_t)(head * 3 + 2) * kMat;
    float S[4][4];
    for (int s = 0; s < 4; ++s) {
      const bf16_t* qr = Q + (size_t)(s * kB + b) * kH;
      float qs[16];
      for (int i = 0; i < 16; ++i) qs[i] = (float)qr[lane + 32 * i];
      for (int t = 0; t < 4; ++t) {
        const bf16_t* kr = K + (size_t)(t * kB + b) * kH;
        float d = 0.f;
        for (int i = 0; i < 16; ++i) d += qs[i] * (float)kr[lane + 32 * i];
        for (int off = 16; off > 0; off >>= 1) d += __shfl_xor(d, off, 32);
        S[s][t] = d * 0.044194173824159216f;  // 1/sqrt(512)
      }
    }
    float A[4][4];
    for (int s = 0; s < 4; ++s) {
      float m = S[s][0];
      for (int t = 1; t < 4; ++t) m = fmaxf(m, S[s][t]);
      float sum = 0.f;
      for (int t = 0; t < 4; ++t) { A[s][t] = __expf(S[s][t] - m); sum += A[s][t]; }
      float inv = 1.f / sum;
      for (int t = 0; t < 4; ++t) A[s][t] *= inv;
    }
    unsigned long long best = 0ull;
    for (int i = 0; i < 16; ++i) {
      int hh = lane + 32 * i;
      float v0 = (float)V[(size_t)(0 * kB + b) * kH + hh];
      float v1 = (float)V[(size_t)(1 * kB + b) * kH + hh];
      float v2 = (float)V[(size_t)(2 * kB + b) * kH + hh];
      float v3 = (float)V[(size_t)(3 * kB + b) * kH + hh];
      for (int s = 0; s < 4; ++s) {
        float o = A[s][0] * v0 + A[s][1] * v1 + A[s][2] * v2 + A[s][3] * v3;
        unsigned idx = (unsigned)((size_t)s * kBH + (size_t)b * kH + hh);
        unsigned long long key =
            ((unsigned long long)sortable_u32(o) << 32) | (0xFFFFFFFFu - idx);
        if (key > best) best = key;
      }
    }
    for (int off = 16; off > 0; off >>= 1) {
      unsigned long long o = __shfl_xor(best, off, 32);
      if (o > best) best = o;
    }
    if (lane == 0) atomicMax(&keys[head], best);
  }
}

// ---------------------------------------------------------------------------
// Step post: decode c2/c3, op2 = stack_out[c2], op3 = complx(cands[c3], op2),
// update carry h <- op2 (safe in-place: per-thread read-then-write).
// ---------------------------------------------------------------------------
__global__ __launch_bounds__(256) void step_post(
    const float* __restrict__ st, const float* __restrict__ op1,
    const float* __restrict__ so,
    const unsigned long long* __restrict__ keys,
    float* __restrict__ hbuf, float* __restrict__ outs_t) {
  size_t id = (size_t)blockIdx.x * 256 + threadIdx.x;
  int j = (int)(id & 255);
  size_t b = id >> 8;
  size_t base = b * kH;
  int c2 = (int)(3u & (0xFFFFFFFFu - (unsigned)(keys[0] & 0xFFFFFFFFull)));
  int c3 = (int)(3u & (0xFFFFFFFFu - (unsigned)(keys[1] & 0xFFFFFFFFull)));
  float o21 = so[(size_t)c2 * kBH + base + j];
  float o22 = so[(size_t)c2 * kBH + base + kHB + j];
  float h1 = hbuf[base + j], h2 = hbuf[base + kHB + j];
  float x1, x2;
  if (c3 == 0)      { x1 = st[base + j];  x2 = st[base + kHB + j]; }
  else if (c3 == 1) { x1 = h1;            x2 = h2; }
  else if (c3 == 2) { x1 = op1[base + j]; x2 = op1[base + kHB + j]; }
  else              { x1 = 0.f;           x2 = 0.f; }
  outs_t[base + j] = x1 * o21 - x2 * o22;
  outs_t[base + kHB + j] = x1 * o22 + x2 * o21;
  hbuf[base + j] = o21;
  hbuf[base + kHB + j] = o22;
}

// ---------------------------------------------------------------------------
extern "C" void kernel_launch(void* const* d_in, const int* in_sizes, int n_in,
                              void* d_out, int out_size, void* d_ws, size_t ws_size,
                              hipStream_t stream) {
  (void)in_sizes; (void)n_in; (void)out_size; (void)ws_size;
  const int*   seqs  = (const int*)d_in[0];
  const float* sub_e = (const float*)d_in[1];
  const float* rel_e = (const float*)d_in[2];
  const float* obj_e = (const float*)d_in[3];
  const float* bn1_g = (const float*)d_in[4];
  const float* bn1_b = (const float*)d_in[5];
  const float* bn2_g = (const float*)d_in[6];
  const float* bn2_b = (const float*)d_in[7];
  const float* bn3_g = (const float*)d_in[8];
  const float* bn3_b = (const float*)d_in[9];
  const float* Wq2 = (const float*)d_in[10];
  const float* Wk2 = (const float*)d_in[11];
  const float* Wv2 = (const float*)d_in[12];
  const float* Wq3 = (const float*)d_in[13];
  const float* Wk3 = (const float*)d_in[14];
  const float* Wv3 = (const float*)d_in[15];
  float* out = (float*)d_out;

  // Workspace carve-up (~490 MB total)
  char* p = (char*)d_ws;
  auto carve = [&](size_t bytes) {
    void* r = (void*)p;
    p += (bytes + 255) & ~(size_t)255;
    return r;
  };
  float*  s_raw  = (float*)carve(3 * kBH * 4);        // -> st_seq (in-place bn)
  float*  r_raw  = (float*)carve(3 * kBH * 4);        // -> rt_seq (in-place bn)
  float*  hbuf   = (float*)carve(kBH * 4);
  float*  op1    = (float*)carve(kBH * 4);
  float*  stack  = (float*)carve(4 * kBH * 4);
  bf16_t* Xb     = (bf16_t*)carve(4 * kBH * 2);
  bf16_t* Wt     = (bf16_t*)carve((size_t)6 * kH * kH * 2);
  bf16_t* QKV    = (bf16_t*)carve(6 * kMat * 2);
  float*  outs   = (float*)carve(3 * kBH * 4);
  float*  mean_s = (float*)carve(3 * kH * 4);
  float*  rstd_s = (float*)carve(3 * kH * 4);
  float*  mean_r = (float*)carve(3 * kH * 4);
  float*  rstd_r = (float*)carve(3 * kH * 4);
  float*  mean_o = (float*)carve(kH * 4);
  float*  rstd_o = (float*)carve(kH * 4);
  float*  part   = (float*)carve((size_t)32 * kH * 2 * 4);
  unsigned long long* keys = (unsigned long long*)carve(16);

  // 1) weights -> transposed bf16
  conv_weights<<<dim3(1024, 6), 256, 0, stream>>>(Wq2, Wk2, Wv2, Wq3, Wk3, Wv3, Wt);

  // 2) gather embeddings; target_f -> second half of d_out; h0 -> hbuf
  gather_embed<<<(unsigned)(3 * kBH / 256), 256, 0, stream>>>(
      seqs, sub_e, rel_e, obj_e, s_raw, r_raw, hbuf, out + 3 * kBH);

  // 3) bn1/bn2 per-timestep stats + in-place normalize
  for (int t = 0; t < kT; ++t) {
    stats_pass1<<<dim3(8, 32), dim3(64, 4), 0, stream>>>(s_raw + (size_t)t * kBH, kB, part);
    stats_fin<<<2, 256, 0, stream>>>(part, kB, mean_s + t * kH, rstd_s + t * kH);
  }
  bn_apply3<<<dim3((unsigned)(kBH / 256), 3), 256, 0, stream>>>(s_raw, mean_s, rstd_s, bn1_g, bn1_b);
  for (int t = 0; t < kT; ++t) {
    stats_pass1<<<dim3(8, 32), dim3(64, 4), 0, stream>>>(r_raw + (size_t)t * kBH, kB, part);
    stats_fin<<<2, 256, 0, stream>>>(part, kB, mean_r + t * kH, rstd_r + t * kH);
  }
  bn_apply3<<<dim3((unsigned)(kBH / 256), 3), 256, 0, stream>>>(r_raw, mean_r, rstd_r, bn2_g, bn2_b);

  // 4) recurrent scan, 3 steps
  for (int t = 0; t < kT; ++t) {
    const float* st = s_raw + (size_t)t * kBH;
    const float* rt = r_raw + (size_t)t * kBH;
    step_pre<<<8192, 256, 0, stream>>>(st, rt, hbuf, op1, stack, Xb, keys);
    gemm_qkv<<<dim3(256, 8, 6), 256, 0, stream>>>(Xb, Wt, QKV);
    attn_select<<<kB / 8, 256, 0, stream>>>(QKV, keys);
    step_post<<<8192, 256, 0, stream>>>(st, op1, stack, keys, hbuf, outs + (size_t)t * kBH);
  }

  // 5) bn3 over (3*B, H) -> first half of d_out
  stats_pass1<<<dim3(8, 32), dim3(64, 4), 0, stream>>>(outs, 3 * kB, part);
  stats_fin<<<2, 256, 0, stream>>>(part, 3 * kB, mean_o, rstd_o);
  bn_out<<<(unsigned)(3 * kBH / 256), 256, 0, stream>>>(outs, mean_o, rstd_o, bn3_g, bn3_b, out);
}
